// CascadeCornerPooling_27951647162859
// MI455X (gfx1250) — compile-verified, matrix operand order checked by reading, not run on hardware
//
#include <hip/hip_runtime.h>

typedef __bf16 bf16;
typedef __attribute__((ext_vector_type(8)))  __bf16 v8bf;
typedef __attribute__((ext_vector_type(16))) __bf16 v16bf;
typedef __attribute__((ext_vector_type(8)))  float  v8f;

#define DEVINL __device__ __forceinline__

static DEVINL v16bf cat16(v8bf lo, v8bf hi) {
  return __builtin_shufflevector(lo, hi, 0,1,2,3,4,5,6,7,8,9,10,11,12,13,14,15);
}

// ---------------------------------------------------------------------------
// BN fold: scale = gamma * rsqrt(var+eps), bias = beta - mean*scale
// ---------------------------------------------------------------------------
__global__ void bn_prep_kernel(const float* __restrict__ g, const float* __restrict__ b,
                               const float* __restrict__ m, const float* __restrict__ v,
                               float* __restrict__ scale, float* __restrict__ bias) {
  int c = threadIdx.x;
  if (c < 128) {
    float inv = g[c] * rsqrtf(v[c] + 1e-5f);
    scale[c] = inv;
    bias[c]  = b[c] - m[c] * inv;
  }
}

// ---------------------------------------------------------------------------
// Weight pack: OIHW f32 -> bf16 [tap][chunk][co(128)][ci(32)]
// ---------------------------------------------------------------------------
__global__ void pack_weights_kernel(const float* __restrict__ w, bf16* __restrict__ dst, int CIN) {
  int total = 128 * CIN * 9;
  int nch = CIN >> 5;
  for (int i = blockIdx.x * blockDim.x + threadIdx.x; i < total; i += gridDim.x * blockDim.x) {
    int cis   = i & 31;
    int co    = (i >> 5) & 127;
    int tc    = i >> 12;            // tap*nch + chunk
    int tap   = tc / nch;
    int chunk = tc - tap * nch;
    int ci    = chunk * 32 + cis;
    int kh    = tap / 3;
    int kw    = tap - kh * 3;
    dst[i] = (bf16)w[((size_t)(co * CIN + ci) * 3 + kh) * 3 + kw];
  }
}

// ---------------------------------------------------------------------------
// Layout convert: NCHW f32 -> NHWC bf16 (LDS transpose, coalesced both sides)
// one block per (n,h); H=W=128
// ---------------------------------------------------------------------------
__global__ void to_nhwc_bf16_kernel(const float* __restrict__ src, bf16* __restrict__ dst, int C) {
  const int n = blockIdx.x >> 7;
  const int h = blockIdx.x & 127;
  __shared__ __align__(16) bf16 t[128 * 32];  // [w][ci]
  for (int c0 = 0; c0 < C; c0 += 32) {
    __syncthreads();
    for (int idx = threadIdx.x; idx < 32 * 128; idx += 256) {
      int ci = idx >> 7;
      int w  = idx & 127;
      float v = src[(((size_t)n * C + c0 + ci) * 128 + h) * 128 + w];
      t[w * 32 + ci] = (bf16)v;
    }
    __syncthreads();
    for (int idx = threadIdx.x; idx < 512; idx += 256) {   // 128*32/8 b128 stores
      int w = idx >> 2;
      int q = idx & 3;
      *(v8bf*)(dst + ((size_t)((n * 128 + h) * 128 + w)) * C + c0 + q * 8) =
          *(const v8bf*)(&t[w * 32 + q * 8]);
    }
  }
}

// ---------------------------------------------------------------------------
// Implicit-GEMM 3x3 conv + folded BN (+ReLU), bf16 WMMA, f32 accumulate.
// Input NHWC bf16 [16][128][128][CIN]; output NCHW f32 [16][128][128][128].
// One block per (n,h): full 128(Cout) x 128(W) tile; 8 waves, each wave:
// M = 16 rows, N = 128 (8 tiles of 16). DUAL computes a second conv sharing
// the same input tile and B fragments.
//
// Global->LDS staging uses CDNA5 async copies (GLOBAL_LOAD_ASYNC_TO_LDS_B128,
// ASYNCcnt) with double-buffered LDS: the next ci-chunk's tile streams in
// while the current chunk's WMMA chain executes. Halo is pre-zeroed once
// (OOB pattern is invariant per block), so the steady-state loop issues only
// async copies + one s_wait_asynccnt + one barrier per chunk.
// ---------------------------------------------------------------------------
template <int CIN, bool DUAL, bool RELU>
__launch_bounds__(256)
__global__ void conv3x3_bn_wmma(
    const bf16* __restrict__ xin,
    const bf16* __restrict__ Ap0, const float* __restrict__ scale0,
    const float* __restrict__ bias0, float* __restrict__ out0,
    const bf16* __restrict__ Ap1, const float* __restrict__ scale1,
    const float* __restrict__ bias1, float* __restrict__ out1) {
  constexpr int NCH   = CIN / 32;
  constexpr int TILE  = 3 * 130 * 32;      // bf16 elements per buffer
  const int n = blockIdx.x >> 7;
  const int h = blockIdx.x & 127;
  const int lane = threadIdx.x & 31;
  const int laneN = lane & 15;
  const int hiHalf = lane >> 4;   // 0 or 1
  const int kOct = hiHalf << 3;   // K-octet base: 0 or 8
  const int m_base = (threadIdx.x >> 5) * 16;

  __shared__ __align__(16) bf16 xs[2 * TILE];  // double-buffered [row][wpos][ci]

  // ---- zero both buffers once (halo + OOB rows stay zero forever) ----
  {
    const __bf16 zz = (__bf16)0.0f;
    v8bf vz = {zz, zz, zz, zz, zz, zz, zz, zz};
    for (int idx = threadIdx.x * 8; idx < 2 * TILE; idx += 256 * 8)
      *(v8bf*)(&xs[idx]) = vz;
  }
  __syncthreads();

  // ---- async prefetch of one ci-chunk into buffer `buf` ----
  auto prefetch = [&](int chunk, int buf) {
    const unsigned ldsBase = (unsigned)(uintptr_t)(&xs[buf * TILE]);
    // interior only: 3 rows x 128 w x 4 ci-octets = 1536 items, 6 per thread
    for (int idx = threadIdx.x; idx < 3 * 128 * 4; idx += 256) {
      const int r  = idx >> 9;         // 0..2
      const int rm = idx & 511;
      const int pi = rm >> 2;          // 0..127 (wpos = pi+1)
      const int q  = rm & 3;
      const int hs = h + r - 1;
      if ((unsigned)hs < 128u) {
        unsigned lds_off = ldsBase + (unsigned)(((r * 130 + pi + 1) * 32 + q * 8) * 2);
        unsigned long long ga = (unsigned long long)(uintptr_t)(
            xin + (size_t)((n * 128 + hs) * 128 + pi) * CIN + chunk * 32 + q * 8);
        asm volatile("global_load_async_to_lds_b128 %0, %1, off"
                     :: "v"(lds_off), "v"(ga) : "memory");
      }
    }
  };

  v8f acc0[8], acc1[8];
  {
    v8f z8 = {0.f, 0.f, 0.f, 0.f, 0.f, 0.f, 0.f, 0.f};
#pragma unroll
    for (int i = 0; i < 8; ++i) { acc0[i] = z8; acc1[i] = z8; }
  }

  prefetch(0, 0);

  for (int chunk = 0; chunk < NCH; ++chunk) {
    const int cur = chunk & 1;
    // my async copies into buffer `cur` have landed in LDS:
    asm volatile("s_wait_asynccnt 0x0" ::: "memory");
    // everyone's copies landed + everyone done reading buffer cur^1:
    __syncthreads();
    if (chunk + 1 < NCH) prefetch(chunk + 1, cur ^ 1);

    const bf16* xb = &xs[cur * TILE];

    for (int tap = 0; tap < 9; ++tap) {
      const int kh = tap / 3;
      const int kw = tap - kh * 3;
      // A fragment(s): 16x32 bf16, lane l<16 -> M=l, K{0..7,16..23}; l>=16 -> K{8..15,24..31}
      const bf16* a0p = Ap0 + (size_t)((tap * NCH + chunk) * 128 + m_base + laneN) * 32;
      v16bf a0 = cat16(*(const v8bf*)(a0p + kOct), *(const v8bf*)(a0p + kOct + 16));
      v16bf a1 = a0;
      if (DUAL) {
        const bf16* a1p = Ap1 + (size_t)((tap * NCH + chunk) * 128 + m_base + laneN) * 32;
        a1 = cat16(*(const v8bf*)(a1p + kOct), *(const v8bf*)(a1p + kOct + 16));
      }
      // B fragments with explicit 1-deep rotation to hide ds latency
      const bf16* brow = &xb[(kh * 130 + laneN + kw) * 32 + kOct];
      v16bf bcur = cat16(*(const v8bf*)brow, *(const v8bf*)(brow + 16));
#pragma unroll
      for (int nt = 0; nt < 8; ++nt) {
        v16bf bnext = bcur;
        if (nt < 7) {
          const bf16* bp = brow + (nt + 1) * (16 * 32);
          bnext = cat16(*(const v8bf*)bp, *(const v8bf*)(bp + 16));
        }
        acc0[nt] = __builtin_amdgcn_wmma_f32_16x16x32_bf16(
            false, a0, false, bcur, (short)0, acc0[nt], false, false);
        if (DUAL)
          acc1[nt] = __builtin_amdgcn_wmma_f32_16x16x32_bf16(
              false, a1, false, bcur, (short)0, acc1[nt], false, false);
        bcur = bnext;
      }
    }
  }

  // epilogue: BN (+ReLU), store NCHW f32.
  // C layout: lanes 0-15: N=lane, M=r; lanes 16-31: N=lane-16, M=r+8
#pragma unroll
  for (int nt = 0; nt < 8; ++nt) {
    const int w = nt * 16 + laneN;
#pragma unroll
    for (int r = 0; r < 8; ++r) {
      const int co = m_base + r + kOct;
      {
        float v = acc0[nt][r] * scale0[co] + bias0[co];
        if (RELU) v = fmaxf(v, 0.0f);
        out0[(size_t)((n * 128 + co) * 128 + h) * 128 + w] = v;
      }
      if (DUAL) {
        float v = acc1[nt][r] * scale1[co] + bias1[co];
        if (RELU) v = fmaxf(v, 0.0f);
        out1[(size_t)((n * 128 + co) * 128 + h) * 128 + w] = v;
      }
    }
  }
}

// ---------------------------------------------------------------------------
// TopPool (reverse cummax along H) + add `down`; merge written over `up`.
// one block per (n,co); 128 threads = W. Coalesced over w, serial over h.
// ---------------------------------------------------------------------------
__global__ void toppool_add_kernel(float* __restrict__ up, const float* __restrict__ down) {
  const size_t base = (size_t)blockIdx.x * (128 * 128);
  const int w = threadIdx.x;
  float m = -INFINITY;
  for (int h = 127; h >= 0; --h) {
    size_t i = base + (size_t)h * 128 + w;
    m = fmaxf(m, up[i]);
    up[i] = m + down[i];
  }
}

// ---------------------------------------------------------------------------
// LeftPool (reverse cummax along W): block per (n,co), 128 threads = w,
// Hillis-Steele reverse max-scan in LDS per h-row.
// ---------------------------------------------------------------------------
__global__ void leftpool_kernel(const float* __restrict__ src, float* __restrict__ dst) {
  const size_t base = (size_t)blockIdx.x * (128 * 128);
  const int w = threadIdx.x;
  __shared__ float s[128];
  for (int h = 0; h < 128; ++h) {
    float v = src[base + (size_t)h * 128 + w];
    __syncthreads();
    s[w] = v;
    __syncthreads();
    for (int off = 1; off < 128; off <<= 1) {
      float o = (w + off < 128) ? s[w + off] : -INFINITY;
      __syncthreads();
      v = fmaxf(v, o);
      s[w] = v;
      __syncthreads();
    }
    dst[base + (size_t)h * 128 + w] = v;
  }
}

// ---------------------------------------------------------------------------
extern "C" void kernel_launch(void* const* d_in, const int* in_sizes, int n_in,
                              void* d_out, int out_size, void* d_ws, size_t ws_size,
                              hipStream_t stream) {
  (void)in_sizes; (void)n_in; (void)out_size; (void)ws_size;
  const float* x      = (const float*)d_in[0];
  const float* w_up   = (const float*)d_in[1];
  const float* up_g   = (const float*)d_in[2];
  const float* up_b   = (const float*)d_in[3];
  const float* up_m   = (const float*)d_in[4];
  const float* up_v   = (const float*)d_in[5];
  const float* w_down = (const float*)d_in[6];
  const float* dn_g   = (const float*)d_in[7];
  const float* dn_b   = (const float*)d_in[8];
  const float* dn_m   = (const float*)d_in[9];
  const float* dn_v   = (const float*)d_in[10];
  const float* w_p    = (const float*)d_in[11];
  const float* p_g    = (const float*)d_in[12];
  const float* p_b    = (const float*)d_in[13];
  const float* p_m    = (const float*)d_in[14];
  const float* p_v    = (const float*)d_in[15];

  char* ws = (char*)d_ws;
  const size_t BIG = (size_t)16 * 128 * 128 * 128 * 4;  // 134,217,728 B

  bf16*  xbf   = (bf16*)ws;                 // x in NHWC bf16; later reused as conv_p f32 out
  float* upbuf = (float*)(ws + BIG);        // up (relu) f32 NCHW; later merge f32
  float* dnbuf = (float*)(ws + 2 * BIG);    // down (relu) f32 NCHW
  bf16*  mgbf  = (bf16*)(ws + 2 * BIG);     // merge NHWC bf16 (reuses dn region after consumed)
  float* convp = (float*)ws;                // conv_p output f32 NCHW (reuses x_bf16 region)

  bf16* ApU = (bf16*)(ws + 3 * BIG);                    // 128*256*9 bf16
  bf16* ApD = ApU + (size_t)128 * 256 * 9;
  bf16* ApP = ApD + (size_t)128 * 256 * 9;              // 128*128*9 bf16
  float* bn = (float*)(ws + 3 * BIG + 2 * (size_t)128 * 256 * 9 * 2 + (size_t)128 * 128 * 9 * 2);
  float* sU = bn;       float* bU = bn + 128;
  float* sD = bn + 256; float* bD = bn + 384;
  float* sP = bn + 512; float* bP = bn + 640;

  // 1) BN folds
  bn_prep_kernel<<<1, 128, 0, stream>>>(up_g, up_b, up_m, up_v, sU, bU);
  bn_prep_kernel<<<1, 128, 0, stream>>>(dn_g, dn_b, dn_m, dn_v, sD, bD);
  bn_prep_kernel<<<1, 128, 0, stream>>>(p_g, p_b, p_m, p_v, sP, bP);

  // 2) Weight packs (bf16, WMMA-A-fragment friendly)
  pack_weights_kernel<<<1152, 256, 0, stream>>>(w_up, ApU, 256);
  pack_weights_kernel<<<1152, 256, 0, stream>>>(w_down, ApD, 256);
  pack_weights_kernel<<<576, 256, 0, stream>>>(w_p, ApP, 128);

  // 3) x: NCHW f32 -> NHWC bf16
  to_nhwc_bf16_kernel<<<16 * 128, 256, 0, stream>>>(x, xbf, 256);

  // 4) Fused dual conv (up & down share input tile + B fragments) + BN + ReLU
  conv3x3_bn_wmma<256, true, true><<<16 * 128, 256, 0, stream>>>(
      xbf, ApU, sU, bU, upbuf, ApD, sD, bD, dnbuf);

  // 5) TopPool(up) + down -> merge (in place over upbuf)
  toppool_add_kernel<<<16 * 128, 128, 0, stream>>>(upbuf, dnbuf);

  // 6) merge: NCHW f32 -> NHWC bf16 (into dn region)
  to_nhwc_bf16_kernel<<<16 * 128, 256, 0, stream>>>(upbuf, mgbf, 128);

  // 7) conv_p + BN (no ReLU)
  conv3x3_bn_wmma<128, false, false><<<16 * 128, 256, 0, stream>>>(
      mgbf, ApP, sP, bP, convp, nullptr, nullptr, nullptr, nullptr);

  // 8) LeftPool -> final output
  leftpool_kernel<<<16 * 128, 128, 0, stream>>>(convp, (float*)d_out);
}
